// CoarseSkeletonHead_18236431139314
// MI455X (gfx1250) — matrix operand order repaired and checked by reading, not compiled, not run on hardware
//
#include <hip/hip_runtime.h>
#include <hip/hip_bf16.h>
#include <cstdint>

// ---------------------------------------------------------------------------
// CoarseSkeletonHead for MI455X (gfx1250, wave32, WMMA bf16 + TDM async tiles)
// ---------------------------------------------------------------------------

#define LATENT 1024
#define HID    512
#define NJ     24
#define NOUT   92     // (NJ-1)*4
#define NPAD   96     // padded to 6 tiles of 16
#define BROWS  32768

typedef __attribute__((ext_vector_type(16))) __bf16        v16bf;
typedef __attribute__((ext_vector_type(8)))  float         v8f;
typedef __attribute__((ext_vector_type(4)))  unsigned int  su4;
typedef __attribute__((ext_vector_type(8)))  unsigned int  su8;

union Frag {
    uint4 q[2];
    v16bf v;
};

__device__ constexpr int PAR[NJ] = {-1,0,0,0,1,2,3,4,5,6,7,8,9,9,9,12,13,14,16,17,18,19,20,21};

__device__ __forceinline__ unsigned short f2bf(float f) {
    unsigned int u = __float_as_uint(f);
    u += 0x7FFFu + ((u >> 16) & 1u);           // round-to-nearest-even
    return (unsigned short)(u >> 16);
}
__device__ __forceinline__ unsigned int pack2bf(float a, float b) {
    return (unsigned int)f2bf(a) | ((unsigned int)f2bf(b) << 16);
}

// --------------------------- prep kernels ----------------------------------
// z [32768][1024] fp32 -> zb bf16 row-major (4 floats / thread)
__global__ void k_prep_zb(const float* __restrict__ z, unsigned int* __restrict__ zb) {
    size_t t = (size_t)blockIdx.x * blockDim.x + threadIdx.x;
    float4 f = reinterpret_cast<const float4*>(z)[t];
    uint2 o;
    o.x = pack2bf(f.x, f.y);
    o.y = pack2bf(f.z, f.w);
    reinterpret_cast<uint2*>(zb)[t] = o;
}

// W1 [K=1024][N=512] fp32 -> W1t [N=512][K=1024] bf16 (transposed)
__global__ void k_prep_w1(const float* __restrict__ W1, unsigned short* __restrict__ W1t) {
    int idx = blockIdx.x * blockDim.x + threadIdx.x;     // 512*1024 threads
    int n = idx >> 10, k = idx & 1023;
    W1t[idx] = f2bf(W1[(size_t)k * HID + n]);
}

// W2 [K=512][N=92] fp32 -> W2t [N=96][K=512] bf16 (transposed, zero padded)
__global__ void k_prep_w2(const float* __restrict__ W2, unsigned short* __restrict__ W2t) {
    int idx = blockIdx.x * blockDim.x + threadIdx.x;     // 96*512 threads
    int n = idx >> 9, k = idx & 511;
    float v = (n < NOUT) ? W2[(size_t)k * NOUT + n] : 0.0f;
    W2t[idx] = f2bf(v);
}

// ----------------------- TDM tile DMA (zb -> LDS) ---------------------------
// 2-D descriptor: tile 32(K) x 128(M) bf16 out of a 32768x1024 bf16 tensor.
__device__ __forceinline__ void tdm_load_tile(const unsigned short* gsrc, unsigned int lds_off) {
    unsigned long long ga = (unsigned long long)(size_t)gsrc;
    su4 g0 = { 1u,                                   // count=1, user mode, no gather
               lds_off,                              // lds_addr (bytes)
               (unsigned int)(ga & 0xFFFFFFFFu),     // global_addr[31:0]
               (unsigned int)((ga >> 32) & 0x1FFFFFFu) | (2u << 30) };  // addr[56:32] | type=2
    su8 g1 = { 0x00010000u,                          // wg_mask=0, data_size=1 (2 bytes)
               (1024u << 16),                        // tensor_dim0 = 1024 (bits 79:48 lo)
               (0x8000u << 16),                      // tensor_dim1 = 32768 (bits 111:80 lo)
               (32u << 16),                          // tensor_dim1 hi=0 | tile_dim0 = 32
               128u,                                 // tile_dim1 = 128, tile_dim2 = 0
               1024u,                                // tensor_dim0_stride = 1024 (lo32)
               0u,                                   // stride0 hi | tensor_dim1_stride lo
               0u };
    asm volatile("tensor_load_to_lds %0, %1" :: "s"(g0), "s"(g1) : "memory");
}

// --------------------------- GEMM1 + SiLU ----------------------------------
// h = silu(zb @ W1 + b1), h stored bf16.  Block tile 128x128, 8 waves (4M x 2N),
// wave tile 32x64 -> 2x4 wmma accumulators.  A tiles double-buffered via TDM.
__global__ __launch_bounds__(256) void k_gemm1(
        const unsigned short* __restrict__ zb,
        const unsigned short* __restrict__ W1t,
        const float* __restrict__ b1,
        unsigned short* __restrict__ h) {
    __shared__ unsigned int As[2][128 * 16];  // two 8KB bf16 tile buffers

    const int tid  = threadIdx.x;
    const int m0   = blockIdx.x * 128;
    const int n0   = blockIdx.y * 128;
    const int wave = tid >> 5;
    const int lane = tid & 31;
    const int wm   = wave & 3;               // 4 M-strips of 32 rows
    const int wn   = wave >> 2;              // 2 N-strips of 64 cols
    const int lmod = lane & 15;
    const int lhi  = lane >> 4;

    v8f acc[2][4] = {};

    const unsigned short* ztile = zb + (size_t)m0 * LATENT;
    const unsigned int lds0 = (unsigned int)(size_t)&As[0][0];
    const unsigned int lds1 = (unsigned int)(size_t)&As[1][0];

    // prologue: DMA first A tile
    if (wave == 0) tdm_load_tile(ztile, lds0);

    for (int k0 = 0; k0 < LATENT; k0 += 32) {
        const int buf = (k0 >> 5) & 1;
        if (wave == 0) __builtin_amdgcn_s_wait_tensorcnt(0);
        __syncthreads();   // LDS tile visible to all waves; prev buf fully consumed
        if (wave == 0 && k0 + 32 < LATENT)
            tdm_load_tile(ztile + k0 + 32, buf ? lds0 : lds1);

        // ---- A fragments from LDS (lanes<16: K0-7/16-23, lanes>=16: K8-15/24-31)
        Frag a[2];
#pragma unroll
        for (int mt = 0; mt < 2; ++mt) {
            int row = wm * 32 + mt * 16 + lmod;
            const unsigned int* p = &As[buf][row * 16 + lhi * 4];
            a[mt].q[0] = *reinterpret_cast<const uint4*>(p);
            a[mt].q[1] = *reinterpret_cast<const uint4*>(p + 8);
        }
        // ---- B fragments from bf16-transposed W1 (L2-hot, two 16B loads per lane)
        Frag b[4];
#pragma unroll
        for (int nt = 0; nt < 4; ++nt) {
            int n = n0 + wn * 64 + nt * 16 + lmod;
            const uint4* p = reinterpret_cast<const uint4*>(
                W1t + (size_t)n * LATENT + k0 + lhi * 8);
            b[nt].q[0] = p[0];
            b[nt].q[1] = p[2];
            if (k0 + 32 < LATENT)
                __builtin_prefetch(W1t + (size_t)n * LATENT + k0 + 32 + lhi * 16, 0, 1);
        }
        // ---- 8 WMMAs
#pragma unroll
        for (int mt = 0; mt < 2; ++mt)
#pragma unroll
            for (int nt = 0; nt < 4; ++nt)
                acc[mt][nt] = __builtin_amdgcn_wmma_f32_16x16x32_bf16(
                    false, a[mt].v, false, b[nt].v,
                    (short)0, acc[mt][nt], false, false);
    }

    // ---- epilogue: bias + SiLU + store bf16
#pragma unroll
    for (int mt = 0; mt < 2; ++mt) {
#pragma unroll
        for (int nt = 0; nt < 4; ++nt) {
            int n = n0 + wn * 64 + nt * 16 + lmod;
            float bias = b1[n];
#pragma unroll
            for (int v = 0; v < 8; ++v) {
                int row = m0 + wm * 32 + mt * 16 + v + 8 * lhi;
                float x = acc[mt][nt][v] + bias;
                float s = x / (1.0f + __expf(-x));
                h[(size_t)row * HID + n] = f2bf(s);
            }
        }
    }
}

// ------------------- GEMM2 + forward-kinematics epilogue -------------------
__global__ __launch_bounds__(256) void k_gemm2_fk(
        const unsigned short* __restrict__ h,
        const unsigned short* __restrict__ W2t,
        const float* __restrict__ b2,
        float* __restrict__ out) {
    __shared__ float rawS[128 * NPAD];      // 48 KB

    const int tid  = threadIdx.x;
    const int m0   = blockIdx.x * 128;
    const int wave = tid >> 5;
    const int lane = tid & 31;
    const int lmod = lane & 15;
    const int lhi  = lane >> 4;
    const int rowbase = wave * 16;

    v8f acc[6] = {};

    for (int k0 = 0; k0 < HID; k0 += 32) {
        Frag a;
        const unsigned short* arow =
            h + (size_t)(m0 + rowbase + lmod) * HID + k0;
        a.q[0] = *reinterpret_cast<const uint4*>(arow + lhi * 8);
        a.q[1] = *reinterpret_cast<const uint4*>(arow + 16 + lhi * 8);

        Frag b[6];
#pragma unroll
        for (int nt = 0; nt < 6; ++nt) {
            const unsigned short* brow =
                W2t + (size_t)(nt * 16 + lmod) * HID + k0 + lhi * 8;
            b[nt].q[0] = *reinterpret_cast<const uint4*>(brow);
            b[nt].q[1] = *reinterpret_cast<const uint4*>(brow + 16);
        }
#pragma unroll
        for (int nt = 0; nt < 6; ++nt)
            acc[nt] = __builtin_amdgcn_wmma_f32_16x16x32_bf16(
                false, a.v, false, b[nt].v, (short)0, acc[nt], false, false);
    }

    // ---- dump raw scores (+bias) to LDS
#pragma unroll
    for (int nt = 0; nt < 6; ++nt) {
        int col = nt * 16 + lmod;
        float bias = (col < NOUT) ? b2[col] : 0.0f;
#pragma unroll
        for (int v = 0; v < 8; ++v) {
            int r = rowbase + v + 8 * lhi;
            rawS[r * NPAD + col] = acc[nt][v] + bias;
        }
    }
    __syncthreads();

    // ---- forward kinematics, one lane per row
    if (tid < 128) {
        const float* r = &rawS[tid * NPAD];
        size_t grow = (size_t)m0 + tid;

        float ox[NJ], oy[NJ], oz[NJ], len[NJ - 1];
        ox[0] = oy[0] = oz[0] = 0.0f;
#pragma unroll
        for (int j = 0; j < NJ - 1; ++j) {
            float vx = r[j * 4 + 0], vy = r[j * 4 + 1], vz = r[j * 4 + 2], vw = r[j * 4 + 3];
            float nrm = sqrtf(vx * vx + vy * vy + vz * vz);
            float inv = 1.0f / fmaxf(nrm, 1e-12f);
            float L = (vw > 20.0f) ? vw : log1pf(__expf(vw));
            len[j] = L;
            float s = inv * L;
            ox[j + 1] = vx * s; oy[j + 1] = vy * s; oz[j + 1] = vz * s;
        }
        float jx[NJ], jy[NJ], jz[NJ];
        jx[0] = jy[0] = jz[0] = 0.0f;
#pragma unroll
        for (int j = 1; j < NJ; ++j) {
            int p = PAR[j];
            jx[j] = jx[p] + ox[j];
            jy[j] = jy[p] + oy[j];
            jz[j] = jz[p] + oz[j];
        }
        float* joints = out + grow * 72;
        float* offs   = out + (size_t)BROWS * 72 + grow * 72;
        float* lptr   = out + (size_t)BROWS * 144 + grow * 23;
#pragma unroll
        for (int j = 0; j < NJ; ++j) {
            joints[j * 3 + 0] = jx[j]; joints[j * 3 + 1] = jy[j]; joints[j * 3 + 2] = jz[j];
            offs[j * 3 + 0] = ox[j];   offs[j * 3 + 1] = oy[j];   offs[j * 3 + 2] = oz[j];
        }
#pragma unroll
        for (int j = 0; j < NJ - 1; ++j) lptr[j] = len[j];
    }
}

// ---------------------------------------------------------------------------
extern "C" void kernel_launch(void* const* d_in, const int* in_sizes, int n_in,
                              void* d_out, int out_size, void* d_ws, size_t ws_size,
                              hipStream_t stream) {
    (void)in_sizes; (void)n_in; (void)out_size; (void)ws_size;
    const float* z  = (const float*)d_in[0];
    const float* W1 = (const float*)d_in[1];
    const float* b1 = (const float*)d_in[2];
    const float* W2 = (const float*)d_in[3];
    const float* b2 = (const float*)d_in[4];
    float* out = (float*)d_out;

    // workspace layout (bf16): W1t | W2t | h | zb   (~100 MB)
    unsigned short* w1t = (unsigned short*)d_ws;                       // 512*1024
    unsigned short* w2t = w1t + (size_t)HID * LATENT;                  // 96*512
    unsigned short* h   = w2t + (size_t)NPAD * HID;                    // 32768*512
    unsigned short* zb  = h + (size_t)BROWS * HID;                     // 32768*1024

    k_prep_zb<<<(BROWS * (LATENT / 4)) / 256, 256, 0, stream>>>(z, (unsigned int*)zb);
    k_prep_w1<<<(HID * LATENT) / 256, 256, 0, stream>>>(W1, w1t);
    k_prep_w2<<<(NPAD * HID) / 256, 256, 0, stream>>>(W2, w2t);

    dim3 g1(BROWS / 128, HID / 128);
    k_gemm1<<<g1, 256, 0, stream>>>(zb, w1t, b1, h);

    k_gemm2_fk<<<BROWS / 128, 256, 0, stream>>>(h, w2t, b2, out);
}
